// NTXentLoss_26654567039218
// MI455X (gfx1250) — matrix operand order repaired and checked by reading, not compiled, not run on hardware
//
#include <hip/hip_runtime.h>
#include <hip/hip_fp16.h>
#include <math.h>

#define BS       4096
#define NROW     8192          // 2*BS
#define D        256
#define INV_TAU  2.0f          // 1/0.5
#define COLSPLIT 16
#define NT       (NROW / 16 / COLSPLIT)   // 32 column tiles per workgroup
#define LDSS     264           // padded LDS row stride in halfs (528B -> bank spread)

typedef __attribute__((ext_vector_type(16))) _Float16 v16h;
typedef __attribute__((ext_vector_type(8)))  _Float16 v8h;
typedef __attribute__((ext_vector_type(8)))  float    v8f;
typedef int v4i __attribute__((vector_size(16)));   // matches builtin param pointee

typedef __attribute__((address_space(1))) v4i* g_v4i_p;  // global int4*
typedef __attribute__((address_space(3))) v4i* l_v4i_p;  // LDS int4*

union V16 { v16h v; v8h h[2]; };

#if __has_builtin(__builtin_amdgcn_global_load_async_to_lds_b128) && \
    __has_builtin(__builtin_amdgcn_s_wait_asynccnt)
#define HAVE_ASYNC 1
#else
#define HAVE_ASYNC 0
#endif

// ---------------- normalize rows + fp16 copy + inverse norms ----------------
__global__ __launch_bounds__(256) void k_norm(const float* __restrict__ zi,
                                              const float* __restrict__ zj,
                                              _Float16* __restrict__ z16,
                                              float* __restrict__ invn) {
  const int row  = blockIdx.x * 8 + (threadIdx.x >> 5);
  const int lane = threadIdx.x & 31;
  const float* src = (row < BS) ? (zi + (size_t)row * D) : (zj + (size_t)(row - BS) * D);
  float x[8]; float ss = 0.f;
#pragma unroll
  for (int j = 0; j < 8; ++j) { x[j] = src[lane + 32 * j]; ss += x[j] * x[j]; }
#pragma unroll
  for (int m = 16; m >= 1; m >>= 1) ss += __shfl_xor(ss, m, 32);
  const float inv = 1.0f / fmaxf(sqrtf(ss), 1e-12f);
  _Float16* dst = z16 + (size_t)row * D;
#pragma unroll
  for (int j = 0; j < 8; ++j) dst[lane + 32 * j] = (_Float16)(x[j] * inv);
  if (lane == 0) invn[row] = inv;
}

// ---------------- exact fp32 positive-pair similarities ----------------
__global__ __launch_bounds__(256) void k_pos(const float* __restrict__ zi,
                                             const float* __restrict__ zj,
                                             const float* __restrict__ invn,
                                             float* __restrict__ pos) {
  const int row  = blockIdx.x * 8 + (threadIdx.x >> 5);
  const int lane = threadIdx.x & 31;
  const float* a = zi + (size_t)row * D;
  const float* b = zj + (size_t)row * D;
  float d = 0.f;
#pragma unroll
  for (int j = 0; j < 8; ++j) d += a[lane + 32 * j] * b[lane + 32 * j];
#pragma unroll
  for (int m = 16; m >= 1; m >>= 1) d += __shfl_xor(d, m, 32);
  if (lane == 0) pos[row] = d * invn[row] * invn[row + BS] * INV_TAU;
}

// ---------------- stage one 16x256 f16 B-tile into LDS ----------------
__device__ __forceinline__ void stage_tile(const _Float16* __restrict__ z16,
                                           int colBase, _Float16* sbuf, int tid) {
#pragma unroll
  for (int p = 0; p < 2; ++p) {        // 512 x 16B chunks / 256 threads
    const int i   = tid + p * 256;
    const int row = i >> 5;
    const int c   = i & 31;
    const _Float16* g = z16 + (size_t)(colBase + row) * D + c * 8;
    _Float16*       l = sbuf + row * LDSS + c * 8;
#if HAVE_ASYNC
    __builtin_amdgcn_global_load_async_to_lds_b128(
        (g_v4i_p)(void*)g, (l_v4i_p)(void*)l, 0, 0);
#else
    *(v8h*)l = *(const v8h*)g;
#endif
  }
}

// ---------------- fused z@z^T row-sums of exp (minus diagonal) ----------------
__global__ __launch_bounds__(256) void k_simrows(const _Float16* __restrict__ z16,
                                                 float* __restrict__ neg) {
  __shared__ _Float16 smem[2 * 16 * LDSS];
  const int tid  = threadIdx.x;
  const int lane = tid & 31;
  const int wv   = tid >> 5;
  const int l15  = lane & 15;
  const int hi   = lane >> 4;                 // 0 for lanes 0-15, 1 for 16-31
  const int rowBase = blockIdx.x * 256 + wv * 32;   // 32 rows per wave
  const int colOrg  = blockIdx.y * (16 * NT);

  // A fragments for 32 rows x full K=256 (2 tiles x 8 k-chunks), ISA A layout:
  // lanes 0-15 : halfs = K 0-7 | 16-23 ; lanes 16-31 : halfs = K 8-15 | 24-31
  V16 a[2][8];
#pragma unroll
  for (int t = 0; t < 2; ++t) {
    const _Float16* rp = z16 + (size_t)(rowBase + t * 16 + l15) * D;
#pragma unroll
    for (int kk = 0; kk < 8; ++kk) {
      const int s0 = kk * 32 + (hi ? 8 : 0);
      const int s1 = kk * 32 + (hi ? 24 : 16);
      a[t][kk].h[0] = *(const v8h*)(rp + s0);
      a[t][kk].h[1] = *(const v8h*)(rp + s1);
    }
  }

  float rs0[8], rs1[8];
#pragma unroll
  for (int v = 0; v < 8; ++v) { rs0[v] = 0.f; rs1[v] = 0.f; }

  stage_tile(z16, colOrg, smem, tid);         // prefetch tile 0

  for (int ct = 0; ct < NT; ++ct) {
    _Float16* cur = smem + (ct & 1) * (16 * LDSS);
    if (ct + 1 < NT)
      stage_tile(z16, colOrg + (ct + 1) * 16, smem + ((ct + 1) & 1) * (16 * LDSS), tid);
#if HAVE_ASYNC
    if (ct + 1 < NT) __builtin_amdgcn_s_wait_asynccnt(2);   // keep newest tile in flight
    else             __builtin_amdgcn_s_wait_asynccnt(0);
#endif
    __syncthreads();                           // current tile visible to all waves

    // B fragments from LDS, ISA B layout: lanes 0-15 halfs = K 0-7 | 16-23, lanes 16-31 = K 8-15 | 24-31
    const _Float16* bp = cur + l15 * LDSS + hi * 8;
    V16 b[8];
#pragma unroll
    for (int kk = 0; kk < 8; ++kk) {
      b[kk].h[0] = *(const v8h*)(bp + kk * 32);
      b[kk].h[1] = *(const v8h*)(bp + kk * 32 + 16);
    }

    v8f c0 = {}; v8f c1 = {};
#pragma unroll
    for (int kk = 0; kk < 8; ++kk) {
      c0 = __builtin_amdgcn_wmma_f32_16x16x32_f16(false, a[0][kk].v, false, b[kk].v,
                                                  (short)0, c0, false, false);
      c1 = __builtin_amdgcn_wmma_f32_16x16x32_f16(false, a[1][kk].v, false, b[kk].v,
                                                  (short)0, c1, false, false);
    }

    // C layout: element (M = v + 8*hi, N = l15); accumulate exp, skip diagonal
    const int col = colOrg + ct * 16 + l15;
#pragma unroll
    for (int v = 0; v < 8; ++v) {
      const int r0 = rowBase + v + hi * 8;
      const int r1 = r0 + 16;
      const float e0 = __expf(c0[v] * INV_TAU);
      const float e1 = __expf(c1[v] * INV_TAU);
      if (r0 != col) rs0[v] += e0;
      if (r1 != col) rs1[v] += e1;
    }
    __syncthreads();                           // all reads done before buffer reuse
  }

  // reduce partial row sums across the 16-lane halves, one atomic per row
#pragma unroll
  for (int v = 0; v < 8; ++v) {
    float s0 = rs0[v], s1 = rs1[v];
#pragma unroll
    for (int m = 8; m >= 1; m >>= 1) {
      s0 += __shfl_xor(s0, m, 32);
      s1 += __shfl_xor(s1, m, 32);
    }
    if (l15 == 0) {
      unsafeAtomicAdd(neg + rowBase + v + hi * 8, s0);
      unsafeAtomicAdd(neg + rowBase + 16 + v + hi * 8, s1);
    }
  }
}

// ---------------- final loss: mean(log(neg+eps) - pos) ----------------
__global__ __launch_bounds__(256) void k_loss(const float* __restrict__ neg,
                                              const float* __restrict__ pos,
                                              float* __restrict__ out) {
  __shared__ float sm[256];
  float s = 0.f;
  for (int r = threadIdx.x; r < NROW; r += 256) {
    const int pi = (r < BS) ? r : (r - BS);
    s += logf(neg[r] + 1e-8f) - pos[pi];
  }
  sm[threadIdx.x] = s;
  __syncthreads();
  for (int off = 128; off > 0; off >>= 1) {
    if (threadIdx.x < off) sm[threadIdx.x] += sm[threadIdx.x + off];
    __syncthreads();
  }
  if (threadIdx.x == 0) out[0] = sm[0] / (float)NROW;
}

extern "C" void kernel_launch(void* const* d_in, const int* in_sizes, int n_in,
                              void* d_out, int out_size, void* d_ws, size_t ws_size,
                              hipStream_t stream) {
  (void)in_sizes; (void)n_in; (void)out_size; (void)ws_size;
  const float* zi = (const float*)d_in[0];
  const float* zj = (const float*)d_in[1];
  float* out = (float*)d_out;

  char* ws = (char*)d_ws;
  _Float16* z16 = (_Float16*)ws;                           // 8192*256*2 = 4 MiB
  float* invn   = (float*)(ws + (size_t)NROW * D * 2);     // 32 KiB
  float* neg    = invn + NROW;                             // 32 KiB
  float* pos    = neg + NROW;                              // 16 KiB

  (void)hipMemsetAsync(neg, 0, NROW * sizeof(float), stream);
  k_norm<<<NROW / 8, 256, 0, stream>>>(zi, zj, z16, invn);
  k_pos<<<BS / 8, 256, 0, stream>>>(zi, zj, invn, pos);
  k_simrows<<<dim3(NROW / 256, COLSPLIT), 256, 0, stream>>>(z16, neg);
  k_loss<<<1, 256, 0, stream>>>(neg, pos, out);
}
